// RPN_47725676593504
// MI455X (gfx1250) — matrix operand order repaired and checked by reading, not compiled, not run on hardware
//
#include <hip/hip_runtime.h>
#include <hip/hip_bf16.h>

typedef __attribute__((ext_vector_type(16))) _Float16 v16h;
typedef __attribute__((ext_vector_type(8)))  float    v8f;

// ---------------- workspace layout (bytes) ----------------
// wf16:   256*2304 f16
// scores: 8 * 36288 f32   (per b: [p3 27648][p4 6912][p5 1728])
// boxes:  8 * 36288 * 4 f32
// cand scores: 8*3000 f32 ; cand boxes: 8*3000*4 f32
static constexpr size_t OFF_W16 = 0;
static constexpr size_t OFF_SC  = 1179648;
static constexpr size_t OFF_BX  = OFF_SC + (size_t)8 * 36288 * 4;       // 2340864
static constexpr size_t OFF_CS  = OFF_BX + (size_t)8 * 36288 * 16;      // 6985728
static constexpr size_t OFF_CB  = OFF_CS + (size_t)8 * 3000 * 4;        // 7081728

#define SCALE_CLAMP 3.3322045f   // log(224/8)

// ============ kernel 0: stem weights fp32 OIHW -> f16 [m][(tap)*256+ci] ======
__global__ __launch_bounds__(256) void wcvt_kernel(const float* __restrict__ w,
                                                   _Float16* __restrict__ wf) {
  int t = blockIdx.x * 256 + threadIdx.x;          // 256*2304 = 589824 total
  if (t >= 256 * 2304) return;
  int m = t / 2304, k = t % 2304;
  int tap = k >> 8, ci = k & 255;
  int ky = tap / 3, kx = tap - ky * 3;
  wf[t] = (_Float16)w[((size_t)(m * 256 + ci)) * 9 + ky * 3 + kx];
}

// ============ kernel 1: fused stem conv3x3 + ReLU + 1x1 heads + decode =======
union AFrag { uint4 q[2]; v16h h; };
union BFrag { unsigned int u[8]; v16h h; };

__global__ __launch_bounds__(256) void rpn_fused_kernel(
    const float* __restrict__ f3, const float* __restrict__ f4,
    const float* __restrict__ f5, const _Float16* __restrict__ wf16,
    const float* __restrict__ stem_b,
    const float* __restrict__ obj_w, const float* __restrict__ obj_b,
    const float* __restrict__ box_w, const float* __restrict__ box_b,
    float* __restrict__ scores, float* __restrict__ boxes) {
  __shared__ _Float16 sPatch[54 * 260];   // 3x18 patch cells x 256ch (pad 4)
  __shared__ float    sStem[16 * 260];    // 16 pos x 256 ch (pad 4)
  __shared__ float    sHW[15 * 256];      // [3 obj + 12 box] x 256
  __shared__ float    sHO[16 * 16];       // 16 pos x 15 head outputs
  __shared__ float    sBias[256];

  const int t = threadIdx.x;
  int bid = blockIdx.x;
  int b = bid / 768;
  int tt = bid % 768;
  int lev, H, W, tpr, strd;
  const float* feat;
  if (tt < 576)      { lev = 0; H = 96; W = 96; tpr = 6; strd = 8;  feat = f3; }
  else if (tt < 720) { lev = 1; tt -= 576; H = 48; W = 48; tpr = 3; strd = 16; feat = f4; }
  else               { lev = 2; tt -= 720; H = 24; W = 24; tpr = 2; strd = 32; feat = f5; }
  const int ty = tt / tpr;
  const int tx = (tt % tpr) * 16;

  // head weights + stem bias -> LDS
  for (int i = t; i < 3840; i += 256) sHW[i] = (i < 768) ? obj_w[i] : box_w[i - 768];
  if (t < 256) sBias[t] = stem_b[t];

  // stage 3x18x256 input patch (zero-padded) as f16
  for (int e = t; e < 13824; e += 256) {
    int ci = e / 54, p = e % 54;
    int r = p / 18, col = p % 18;
    int y = ty + r - 1, x = tx + col - 1;
    float v = 0.f;
    if (y >= 0 && y < H && x >= 0 && x < W)
      v = feat[((size_t)(b * 256 + ci) * H + y) * W + x];
    sPatch[p * 260 + ci] = (_Float16)v;
  }
  __syncthreads();

  // implicit GEMM: 8 waves x 2 M-tiles, K = 2304 in 72 steps of 32
  const int w  = t >> 5;
  const int L  = t & 31;
  const int n  = L & 15;                 // B/C column = spatial position
  const int hi = (L & 16) ? 1 : 0;
  const int m0 = (2 * w) * 16 + (L & 15);
  const int m1 = m0 + 16;
  __builtin_prefetch(wf16 + (size_t)m0 * 2304, 0, 0);
  __builtin_prefetch(wf16 + (size_t)m1 * 2304, 0, 0);

  v8f acc0 = {0.f, 0.f, 0.f, 0.f, 0.f, 0.f, 0.f, 0.f};
  v8f acc1 = {0.f, 0.f, 0.f, 0.f, 0.f, 0.f, 0.f, 0.f};

  for (int s = 0; s < 72; ++s) {
    const int tap = s >> 3;
    const int ci0 = (s & 7) << 5;
    const int ky = tap / 3, kx = tap - ky * 3;

    // B fragment: lane covers K = (hi?16:0)+0..15 contiguous channels
    BFrag bf;
    const int hw = (ky * 18 + n + kx) * 260 + ci0 + (hi ? 16 : 0);
#pragma unroll
    for (int j = 0; j < 8; ++j)
      bf.u[j] = *(const unsigned int*)&sPatch[hw + 2 * j];

    // A fragments: 2x16B loads per lane per M-tile (layout-matched wf16)
    const size_t base0 = (size_t)m0 * 2304 + 32 * s + (hi ? 16 : 0);
    const size_t base1 = (size_t)m1 * 2304 + 32 * s + (hi ? 16 : 0);
    AFrag a0, a1;
    a0.q[0] = *(const uint4*)(wf16 + base0);
    a0.q[1] = *(const uint4*)(wf16 + base0 + 8);
    a1.q[0] = *(const uint4*)(wf16 + base1);
    a1.q[1] = *(const uint4*)(wf16 + base1 + 8);

    acc0 = __builtin_amdgcn_wmma_f32_16x16x32_f16(false, a0.h, false, bf.h,
                                                  (short)0, acc0, false, false);
    acc1 = __builtin_amdgcn_wmma_f32_16x16x32_f16(false, a1.h, false, bf.h,
                                                  (short)0, acc1, false, false);
  }

  // C layout: VGPR r -> M = r + 8*hi ; N = n. Bias + ReLU -> LDS stem tile.
#pragma unroll
  for (int r = 0; r < 8; ++r) {
    int ch0 = 2 * w * 16 + r + hi * 8;
    int ch1 = ch0 + 16;
    sStem[n * 260 + ch0] = fmaxf(acc0[r] + sBias[ch0], 0.f);
    sStem[n * 260 + ch1] = fmaxf(acc1[r] + sBias[ch1], 0.f);
  }
  __syncthreads();

  // 1x1 heads: 15 outputs x 16 positions, 256-long dots
  if (t < 240) {
    int o = t >> 4, n2 = t & 15;
    float sum = (o < 3) ? obj_b[o] : box_b[o - 3];
    const float* wr = &sHW[o * 256];
    const float* sr = &sStem[n2 * 260];
#pragma unroll 8
    for (int c = 0; c < 256; ++c) sum += wr[c] * sr[c];
    sHO[n2 * 16 + o] = sum;
  }
  __syncthreads();

  // sigmoid + anchor decode + clip, write (b, lev, (y*W+x)*3+a)
  if (t < 48) {
    int n2 = t / 3, a = t - n2 * 3;
    int x = tx + n2, y = ty;
    if (x < W) {
      float logit = sHO[n2 * 16 + a];
      float score = 1.f / (1.f + expf(-logit));
      float d0 = sHO[n2 * 16 + 3 + a * 4 + 0];
      float d1 = sHO[n2 * 16 + 3 + a * 4 + 1];
      float d2 = sHO[n2 * 16 + 3 + a * 4 + 2];
      float d3 = sHO[n2 * 16 + 3 + a * 4 + 3];
      float base = (float)strd * 8.f;
      float pw, ph;
      if (a == 0)      { pw = base * 1.41421356f; ph = base * 0.70710678f; }
      else if (a == 1) { pw = base;               ph = base; }
      else             { pw = base * 0.70710678f; ph = base * 1.41421356f; }
      float px = ((float)x + 0.5f) * (float)strd;
      float py = ((float)y + 0.5f) * (float)strd;
      float tw = fminf(d2, SCALE_CLAMP), th = fminf(d3, SCALE_CLAMP);
      float cx = px + pw * d0, cy = py + ph * d1;
      float bw = pw * expf(tw), bh = ph * expf(th);
      float x0 = fminf(fmaxf(cx - 0.5f * bw, 0.f), 768.f);
      float y0 = fminf(fmaxf(cy - 0.5f * bh, 0.f), 768.f);
      float x1 = fminf(fmaxf(cx + 0.5f * bw, 0.f), 768.f);
      float y1 = fminf(fmaxf(cy + 0.5f * bh, 0.f), 768.f);
      const int lvOff = (lev == 0) ? 0 : (lev == 1) ? 27648 : 34560;
      size_t idx = (size_t)b * 36288 + lvOff + ((size_t)y * W + x) * 3 + a;
      scores[idx] = score;
      boxes[idx * 4 + 0] = x0;
      boxes[idx * 4 + 1] = y0;
      boxes[idx * 4 + 2] = x1;
      boxes[idx * 4 + 3] = y1;
    }
  }
}

// ============ kernel 2: per (b,level) radix-select top-1000 ==================
__global__ __launch_bounds__(1024) void topk_kernel(const float* __restrict__ scores,
                                                    const float* __restrict__ boxes,
                                                    float* __restrict__ cs,
                                                    float* __restrict__ cb) {
  __shared__ int hist[256];
  __shared__ unsigned sPrefix;
  __shared__ int sKr, cGt, cEq;
  const int bid = blockIdx.x;
  const int b = bid / 3, lev = bid % 3;
  const int Ns[3]   = {27648, 6912, 1728};
  const int offs[3] = {0, 27648, 34560};
  const int N = Ns[lev];
  const float* sc = scores + (size_t)b * 36288 + offs[lev];
  const float* bx = boxes + ((size_t)b * 36288 + offs[lev]) * 4;
  const int t = threadIdx.x;

  if (t == 0) { sPrefix = 0u; sKr = 1000; }
  for (int shift = 24; shift >= 0; shift -= 8) {
    for (int i = t; i < 256; i += 1024) hist[i] = 0;
    __syncthreads();
    unsigned pfx = sPrefix;
    for (int i = t; i < N; i += 1024) {
      unsigned u = __float_as_uint(sc[i]);   // positive floats: order == uint order
      if (shift == 24 || (u >> (shift + 8)) == pfx)
        atomicAdd(&hist[(u >> shift) & 255], 1);
    }
    __syncthreads();
    if (t == 0) {
      int kr = sKr, cum = 0, sel = 0;
      for (int bin = 255; bin >= 0; --bin) {
        cum += hist[bin];
        if (cum >= kr) { sel = bin; sKr = kr - (cum - hist[bin]); break; }
      }
      sPrefix = (pfx << 8) | (unsigned)sel;
    }
    __syncthreads();
  }
  const unsigned T = sPrefix;
  const int kEq = sKr, nGt = 1000 - kEq;
  if (t == 0) { cGt = 0; cEq = 0; }
  __syncthreads();

  float* csb = cs + (size_t)b * 3000 + lev * 1000;
  float* cbb = cb + ((size_t)b * 3000 + lev * 1000) * 4;
  for (int i = t; i < N; i += 1024) {
    unsigned u = __float_as_uint(sc[i]);
    int slot = -1;
    if (u > T) slot = atomicAdd(&cGt, 1);
    else if (u == T) {
      int e = atomicAdd(&cEq, 1);
      if (e < kEq) slot = nGt + e;
    }
    if (slot >= 0) {
      csb[slot] = sc[i];
      cbb[slot * 4 + 0] = bx[i * 4 + 0];
      cbb[slot * 4 + 1] = bx[i * 4 + 1];
      cbb[slot * 4 + 2] = bx[i * 4 + 2];
      cbb[slot * 4 + 3] = bx[i * 4 + 3];
    }
  }
}

// ============ kernel 3: per-batch sort-3000 + greedy NMS + top-300 ===========
__global__ __launch_bounds__(1024) void nms_kernel(const float* __restrict__ cs,
                                                   const float* __restrict__ cb,
                                                   float* __restrict__ out) {
  extern __shared__ char smem[];
  float* key = (float*)smem;                 // 4096
  int*   sid = (int*)(key + 4096);           // 4096
  float* bx  = (float*)(sid + 4096);         // 4096*4
  int*   keep = (int*)(bx + 4096 * 4);       // 4096
  int*   scan = keep + 4096;                 // 4096
  const int b = blockIdx.x, t = threadIdx.x;
  const float* csb = cs + (size_t)b * 3000;
  const float* cbb = cb + (size_t)b * 3000 * 4;

  for (int i = t; i < 4096; i += 1024) {
    key[i] = (i < 3000) ? csb[i] : -3.4e38f;
    sid[i] = i;
  }
  __syncthreads();

  // bitonic sort descending on (key, sid)
  for (int k = 2; k <= 4096; k <<= 1) {
    for (int j = k >> 1; j > 0; j >>= 1) {
      for (int i = t; i < 4096; i += 1024) {
        int l = i ^ j;
        if (l > i) {
          bool descBlock = ((i & k) == 0);
          float ki = key[i], kl = key[l];
          if ((ki < kl) == descBlock) {
            key[i] = kl; key[l] = ki;
            int s = sid[i]; sid[i] = sid[l]; sid[l] = s;
          }
        }
      }
      __syncthreads();
    }
  }

  // gather boxes into sorted order
  for (int i = t; i < 4096; i += 1024) {
    int s = sid[i];
    if (i < 3000 && s < 3000) {
      bx[i * 4 + 0] = cbb[s * 4 + 0];
      bx[i * 4 + 1] = cbb[s * 4 + 1];
      bx[i * 4 + 2] = cbb[s * 4 + 2];
      bx[i * 4 + 3] = cbb[s * 4 + 3];
    } else {
      bx[i * 4 + 0] = 0.f; bx[i * 4 + 1] = 0.f;
      bx[i * 4 + 2] = 0.f; bx[i * 4 + 3] = 0.f;
    }
    keep[i] = (i < 3000) ? 1 : 0;
  }
  __syncthreads();

  // greedy NMS
  for (int i = 0; i < 3000; ++i) {
    if (keep[i]) {
      float x0 = bx[i * 4], y0 = bx[i * 4 + 1];
      float x1 = bx[i * 4 + 2], y1 = bx[i * 4 + 3];
      float ai = (x1 - x0) * (y1 - y0);
      for (int jj = i + 1 + t; jj < 3000; jj += 1024) {
        if (keep[jj]) {
          float u0 = fmaxf(x0, bx[jj * 4]);
          float v0 = fmaxf(y0, bx[jj * 4 + 1]);
          float u1 = fminf(x1, bx[jj * 4 + 2]);
          float v1 = fminf(y1, bx[jj * 4 + 3]);
          float inter = fmaxf(u1 - u0, 0.f) * fmaxf(v1 - v0, 0.f);
          float aj = (bx[jj * 4 + 2] - bx[jj * 4]) * (bx[jj * 4 + 3] - bx[jj * 4 + 1]);
          float iou = inter / (ai + aj - inter);
          if (iou > 0.7f) keep[jj] = 0;
        }
      }
    }
    __syncthreads();
  }

  // inclusive scan of keep (Hillis-Steele, 4 elems/thread)
  for (int i = t; i < 4096; i += 1024) scan[i] = keep[i];
  __syncthreads();
  for (int off = 1; off < 4096; off <<= 1) {
    int v[4];
    int c = 0;
    for (int i = t; i < 4096; i += 1024, ++c) v[c] = (i >= off) ? scan[i - off] : 0;
    __syncthreads();
    c = 0;
    for (int i = t; i < 4096; i += 1024, ++c) scan[i] += v[c];
    __syncthreads();
  }
  const int K = scan[2999];
  float* ob = out + (size_t)b * 1200;
  for (int i = t; i < 3000; i += 1024) {
    int slot = keep[i] ? (scan[i] - 1) : (K + i - scan[i]);
    if (slot < 300) {
      ob[slot * 4 + 0] = bx[i * 4 + 0];
      ob[slot * 4 + 1] = bx[i * 4 + 1];
      ob[slot * 4 + 2] = bx[i * 4 + 2];
      ob[slot * 4 + 3] = bx[i * 4 + 3];
    }
  }
}

// ================================ launch =====================================
extern "C" void kernel_launch(void* const* d_in, const int* in_sizes, int n_in,
                              void* d_out, int out_size, void* d_ws, size_t ws_size,
                              hipStream_t stream) {
  (void)in_sizes; (void)n_in; (void)out_size; (void)ws_size;
  const float* f3     = (const float*)d_in[0];
  const float* f4     = (const float*)d_in[1];
  const float* f5     = (const float*)d_in[2];
  const float* stem_w = (const float*)d_in[3];
  const float* stem_b = (const float*)d_in[4];
  const float* obj_w  = (const float*)d_in[5];
  const float* obj_b  = (const float*)d_in[6];
  const float* box_w  = (const float*)d_in[7];
  const float* box_b  = (const float*)d_in[8];

  char* ws = (char*)d_ws;
  _Float16* wf16 = (_Float16*)(ws + OFF_W16);
  float* scores  = (float*)(ws + OFF_SC);
  float* boxes   = (float*)(ws + OFF_BX);
  float* cscore  = (float*)(ws + OFF_CS);
  float* cbox    = (float*)(ws + OFF_CB);

  wcvt_kernel<<<2304, 256, 0, stream>>>(stem_w, wf16);
  rpn_fused_kernel<<<8 * 768, 256, 0, stream>>>(f3, f4, f5, wf16, stem_b,
                                                obj_w, obj_b, box_w, box_b,
                                                scores, boxes);
  topk_kernel<<<24, 1024, 0, stream>>>(scores, boxes, cscore, cbox);

  const int nmsSmem = 4096 * 32;  // 128 KB dynamic LDS (<< 320 KB/WGP)
  hipFuncSetAttribute((const void*)nms_kernel,
                      hipFuncAttributeMaxDynamicSharedMemorySize, nmsSmem);
  nms_kernel<<<8, 1024, nmsSmem, stream>>>(cscore, cbox, (float*)d_out);
}
